// LayoutLMv2SelfAttention_16028817949014
// MI455X (gfx1250) — matrix operand order repaired and checked by reading, not compiled, not run on hardware
//
#include <hip/hip_runtime.h>

#define S_ 1024
#define D_ 1024
#define H_ 16
#define DH_ 64
#define B_ 4
#define N3 3072
#define NEG_INF (-__builtin_inff())

typedef __attribute__((ext_vector_type(16))) __bf16          v16bf;
typedef __attribute__((ext_vector_type(8)))  float           v8f;
typedef __attribute__((ext_vector_type(8)))  unsigned short  ushort8;

union FragU {
    struct { ushort8 lo; ushort8 hi; } s;
    v16bf v;
};

__device__ __forceinline__ unsigned short f2bf(float f) {
    union { float f; unsigned int u; } c; c.f = f;
    unsigned int u = c.u;
    u += 0x7FFFu + ((u >> 16) & 1u);          // round-to-nearest-even
    return (unsigned short)(u >> 16);
}

__device__ __forceinline__ v8f wmma_bf16(v16bf a, v16bf b, v8f c) {
    return __builtin_amdgcn_wmma_f32_16x16x32_bf16(false, a, false, b, (short)0, c, false, false);
}

// A-matrix fragment (16x32 bf16): lane holds row m = lane%16.
// K pattern per lane: 16 consecutive at base = (lane>=16)?16:0.
__device__ __forceinline__ v16bf ldA(const unsigned short* row, int koff, int lane) {
    int base = koff + ((lane & 16) ? 16 : 0);
    FragU u;
    u.s.lo = *(const ushort8*)(row + base);
    u.s.hi = *(const ushort8*)(row + base + 8);
    return u.v;
}

// B-matrix fragment (32x16 bf16): lane holds col n = lane%16.
// K pattern per lane: 16 consecutive at base = (lane>=16)?16:0.
__device__ __forceinline__ v16bf ldB(const unsigned short* row, int koff, int lane) {
    int base = koff + ((lane & 16) ? 16 : 0);
    FragU u;
    u.s.lo = *(const ushort8*)(row + base);
    u.s.hi = *(const ushort8*)(row + base + 8);
    return u.v;
}

// ---------------------------------------------------------------------------
// Kernel 1: qkv = hidden @ W  (M=4096, K=1024, N=3072), bf16 WMMA, LDS tiled.
// Epilogue: q -> (q+q_bias)*0.125 into qh[BH,S,DH]; k -> kh[BH,S,DH];
//           v -> (v+v_bias) into vT[BH,DH,S] (transposed for attention B-op).
// ---------------------------------------------------------------------------
__global__ __launch_bounds__(256) void qkv_gemm_kernel(
    const float* __restrict__ hs, const float* __restrict__ W,
    const float* __restrict__ qbias, const float* __restrict__ vbias,
    unsigned short* __restrict__ qws, unsigned short* __restrict__ kws,
    unsigned short* __restrict__ vT)
{
    __shared__ unsigned short As[64][40];   // [m][k] bf16, +8 pad
    __shared__ unsigned short Bs[64][40];   // [n][k] bf16 (transposed), +8 pad

    const int tid  = threadIdx.x;
    const int lane = tid & 31;
    const int wave = tid >> 5;
    const int n0   = blockIdx.x * 64;
    const int m0   = blockIdx.y * 64;
    const int m_off = (wave & 3) * 16;
    const int n_off = (wave >> 2) * 32;
    const int ln    = lane & 15;
    const int half  = lane >> 4;

    v8f acc0 = {}; v8f acc1 = {};

    for (int kk = 0; kk < 32; ++kk) {
        const int k0 = kk * 32;
        {   // A tile: 64 rows x 32 k, coalesced along k
            const int ki = tid & 31, mi = tid >> 5;
            #pragma unroll
            for (int p = 0; p < 8; ++p) {
                const int m = p * 8 + mi;
                As[m][ki] = f2bf(hs[(size_t)(m0 + m) * D_ + k0 + ki]);
            }
        }
        {   // B tile: 32 k x 64 n from W[k][n], stored transposed [n][k]
            const int ni = tid & 63, kb = tid >> 6;
            #pragma unroll
            for (int p = 0; p < 8; ++p) {
                const int k = p * 4 + kb;
                Bs[ni][k] = f2bf(W[(size_t)(k0 + k) * N3 + n0 + ni]);
            }
        }
        __syncthreads();
        v16bf a  = ldA(&As[m_off + ln][0], 0, lane);
        v16bf b0 = ldB(&Bs[n_off + ln][0], 0, lane);
        v16bf b1 = ldB(&Bs[n_off + 16 + ln][0], 0, lane);
        acc0 = wmma_bf16(a, b0, acc0);
        acc1 = wmma_bf16(a, b1, acc1);
        __syncthreads();
    }

    #pragma unroll
    for (int nb = 0; nb < 2; ++nb) {
        v8f acc = nb ? acc1 : acc0;
        const int gn = n0 + n_off + nb * 16 + ln;
        #pragma unroll
        for (int r = 0; r < 8; ++r) {
            const int gm = m0 + m_off + r + 8 * half;
            const int bb = gm >> 10, s = gm & 1023;
            const float c = acc[r];
            if (gn < 1024) {                         // Q
                const int d = gn, h = d >> 6, dh = d & 63;
                qws[((size_t)(bb * H_ + h) * S_ + s) * DH_ + dh] =
                    f2bf((c + qbias[d]) * 0.125f);
            } else if (gn < 2048) {                  // K
                const int d = gn - 1024, h = d >> 6, dh = d & 63;
                kws[((size_t)(bb * H_ + h) * S_ + s) * DH_ + dh] = f2bf(c);
            } else {                                 // V (transposed)
                const int d = gn - 2048, h = d >> 6, dh = d & 63;
                vT[((size_t)(bb * H_ + h) * DH_ + dh) * S_ + s] =
                    f2bf(c + vbias[d]);
            }
        }
    }
}

// ---------------------------------------------------------------------------
// Kernel 2: flash attention, one 16-row q-tile per wave (no LDS).
// S^T = K_tile @ Q^T via WMMA, +rel_pos+rel_2d+mask, online softmax,
// ctx^T = V^T @ P^T via WMMA (P^T register layout == B fragment after one
// cross-half shuffle). Output f32 [B,S,D].
// ---------------------------------------------------------------------------
__global__ __launch_bounds__(256) void attn_kernel(
    const unsigned short* __restrict__ qws, const unsigned short* __restrict__ kws,
    const unsigned short* __restrict__ vT,
    const float* __restrict__ rel_pos, const float* __restrict__ rel2d,
    const int* __restrict__ mask, float* __restrict__ out)
{
    const int lane = threadIdx.x & 31;
    const int wid  = (blockIdx.x << 3) + (threadIdx.x >> 5);  // 4096 waves
    const int qt   = wid & 63;          // q tile (16 rows each)
    const int bh   = wid >> 6;          // fused (b,h), 0..63
    const int b    = bh >> 4;
    const int h    = bh & 15;
    const int n    = lane & 15;
    const int half = lane >> 4;
    const int q    = qt * 16 + n;

    // Q as B-operand: lane holds its q row, 16 consecutive dh per fragment half
    const unsigned short* qrow = qws + ((size_t)bh * S_ + q) * DH_;
    const v16bf qf0 = ldB(qrow, 0, lane);
    const v16bf qf1 = ldB(qrow, 32, lane);

    const float* rp_row = rel_pos + ((size_t)bh * S_ + q) * S_;
    const float* r2_row = rel2d   + ((size_t)bh * S_ + q) * S_;
    const int*   mrow   = mask + b * S_;

    v8f acc[4]; acc[0] = v8f{}; acc[1] = v8f{}; acc[2] = v8f{}; acc[3] = v8f{};
    float m_run = NEG_INF, l_run = 0.0f;

    #pragma unroll 1
    for (int kc = 0; kc < 32; ++kc) {
        const int kb = kc * 32;

        if (kc + 1 < 32) {  // prefetch next chunk's rel rows (global_prefetch_b8)
            __builtin_prefetch(rp_row + kb + 32 + 8 * half, 0, 0);
            __builtin_prefetch(r2_row + kb + 32 + 8 * half, 0, 0);
        }

        // scores^T: rows = keys, cols = q.  Two 16-key tiles.
        const unsigned short* k0row = kws + ((size_t)bh * S_ + kb + n) * DH_;
        const unsigned short* k1row = kws + ((size_t)bh * S_ + kb + 16 + n) * DH_;
        v8f s0 = {}; v8f s1 = {};
        s0 = wmma_bf16(ldA(k0row, 0,  lane), qf0, s0);
        s0 = wmma_bf16(ldA(k0row, 32, lane), qf1, s0);
        s1 = wmma_bf16(ldA(k1row, 0,  lane), qf0, s1);
        s1 = wmma_bf16(ldA(k1row, 32, lane), qf1, s1);

        // add rel_pos + rel_2d_pos, apply mask. s0 keys: kb+8*half+r; s1: +16.
        {
            const int ka = kb + 8 * half;
            const int kc1 = ka + 16;
            float4 ra = *(const float4*)(rp_row + ka);
            float4 rb = *(const float4*)(rp_row + ka + 4);
            float4 ta = *(const float4*)(r2_row + ka);
            float4 tb = *(const float4*)(r2_row + ka + 4);
            s0[0] += ra.x + ta.x; s0[1] += ra.y + ta.y; s0[2] += ra.z + ta.z; s0[3] += ra.w + ta.w;
            s0[4] += rb.x + tb.x; s0[5] += rb.y + tb.y; s0[6] += rb.z + tb.z; s0[7] += rb.w + tb.w;
            ra = *(const float4*)(rp_row + kc1);
            rb = *(const float4*)(rp_row + kc1 + 4);
            ta = *(const float4*)(r2_row + kc1);
            tb = *(const float4*)(r2_row + kc1 + 4);
            s1[0] += ra.x + ta.x; s1[1] += ra.y + ta.y; s1[2] += ra.z + ta.z; s1[3] += ra.w + ta.w;
            s1[4] += rb.x + tb.x; s1[5] += rb.y + tb.y; s1[6] += rb.z + tb.z; s1[7] += rb.w + tb.w;
            int4 ma = *(const int4*)(mrow + ka);
            int4 mb = *(const int4*)(mrow + ka + 4);
            if (ma.x) s0[0] = NEG_INF; if (ma.y) s0[1] = NEG_INF;
            if (ma.z) s0[2] = NEG_INF; if (ma.w) s0[3] = NEG_INF;
            if (mb.x) s0[4] = NEG_INF; if (mb.y) s0[5] = NEG_INF;
            if (mb.z) s0[6] = NEG_INF; if (mb.w) s0[7] = NEG_INF;
            ma = *(const int4*)(mrow + kc1);
            mb = *(const int4*)(mrow + kc1 + 4);
            if (ma.x) s1[0] = NEG_INF; if (ma.y) s1[1] = NEG_INF;
            if (ma.z) s1[2] = NEG_INF; if (ma.w) s1[3] = NEG_INF;
            if (mb.x) s1[4] = NEG_INF; if (mb.y) s1[5] = NEG_INF;
            if (mb.z) s1[6] = NEG_INF; if (mb.w) s1[7] = NEG_INF;
        }

        // online softmax: each lane's 16 values + partner lane (xor 16) = full row
        float cmax = NEG_INF;
        #pragma unroll
        for (int r = 0; r < 8; ++r) { cmax = fmaxf(cmax, s0[r]); cmax = fmaxf(cmax, s1[r]); }
        cmax = fmaxf(cmax, __shfl_xor(cmax, 16, 32));
        const float nm  = fmaxf(m_run, cmax);
        const float nms = (nm == NEG_INF) ? 0.0f : nm;   // -inf-safe
        const float alpha = __expf(m_run - nms);
        float local = 0.0f;
        #pragma unroll
        for (int r = 0; r < 8; ++r) {
            s0[r] = __expf(s0[r] - nms);
            s1[r] = __expf(s1[r] - nms);
            local += s0[r] + s1[r];
        }
        const float rowsum = local + __shfl_xor(local, 16, 32);
        l_run = l_run * alpha + rowsum;
        m_run = nm;
        #pragma unroll
        for (int g = 0; g < 4; ++g)
            #pragma unroll
            for (int r = 0; r < 8; ++r) acc[g][r] *= alpha;

        // Pack P^T as B fragment: lane<16 needs keys 0..15 (own s0 + partner s0),
        // lane>=16 needs keys 16..31 (partner s1 + own s1).
        ushort8 plo, phi;
        #pragma unroll
        for (int r = 0; r < 8; ++r) {
            const float send = half ? s0[r] : s1[r];
            const float recv = __shfl_xor(send, 16, 32);
            const float first = half ? recv : s0[r];
            const float last  = half ? s1[r] : recv;
            plo[r] = f2bf(first);
            phi[r] = f2bf(last);
        }
        FragU pu; pu.s.lo = plo; pu.s.hi = phi;
        const v16bf pB = pu.v;

        // ctx^T += V^T(16d x 32k) @ P^T(32k x 16q)
        #pragma unroll
        for (int g = 0; g < 4; ++g) {
            const unsigned short* vrow = vT + ((size_t)(bh * DH_ + g * 16 + n)) * S_;
            acc[g] = wmma_bf16(ldA(vrow, kb, lane), pB, acc[g]);
        }
    }

    // epilogue: divide by l, write f32 ctx at out[b, q, h*64 + d]
    const float inv = 1.0f / l_run;
    float* orow = out + ((size_t)(b * S_ + q)) * D_ + h * 64;
    #pragma unroll
    for (int g = 0; g < 4; ++g) {
        const int d0 = g * 16 + 8 * half;
        float4 v0 = make_float4(acc[g][0] * inv, acc[g][1] * inv, acc[g][2] * inv, acc[g][3] * inv);
        float4 v1 = make_float4(acc[g][4] * inv, acc[g][5] * inv, acc[g][6] * inv, acc[g][7] * inv);
        *(float4*)(orow + d0)     = v0;
        *(float4*)(orow + d0 + 4) = v1;
    }
}

extern "C" void kernel_launch(void* const* d_in, const int* in_sizes, int n_in,
                              void* d_out, int out_size, void* d_ws, size_t ws_size,
                              hipStream_t stream) {
    const float* hs    = (const float*)d_in[0];   // [B,S,D]
    const int*   mask  = (const int*)  d_in[1];   // [B,1,1,S]
    const float* rp    = (const float*)d_in[2];   // [B,H,S,S]
    const float* r2    = (const float*)d_in[3];   // [B,H,S,S]
    const float* W     = (const float*)d_in[4];   // [D,3D]
    const float* qbias = (const float*)d_in[5];   // [D]
    const float* vbias = (const float*)d_in[6];   // [D]
    float* out = (float*)d_out;                   // [B,S,D]

    const size_t HEAD = (size_t)B_ * H_ * S_ * DH_;   // 4,194,304 elems
    unsigned short* qws = (unsigned short*)d_ws;
    unsigned short* kws = qws + HEAD;
    unsigned short* vT  = kws + HEAD;                 // 24 MB total bf16 staging

    qkv_gemm_kernel<<<dim3(N3 / 64, (B_ * S_) / 64), 256, 0, stream>>>(
        hs, W, qbias, vbias, qws, kws, vT);

    attn_kernel<<<dim3((B_ * H_ * (S_ / 16)) / 8), 256, 0, stream>>>(
        qws, kws, vT, rp, r2, mask, out);
}
